// Attention_79207786873592
// MI455X (gfx1250) — matrix operand order repaired and checked by reading, not compiled, not run on hardware
//
#include <hip/hip_runtime.h>
#include <hip/hip_bf16.h>

#define B_  4
#define L_  2048
#define D_  2048
#define H_  16
#define HD_ 128
#define EPS_ 1e-5f

typedef __attribute__((ext_vector_type(16))) __bf16 v16bf;
typedef __attribute__((ext_vector_type(8)))  __bf16 v8bf;
typedef __attribute__((ext_vector_type(8)))  float  v8f;

static __device__ __forceinline__ v16bf cat8(v8bf lo, v8bf hi) {
  return __builtin_shufflevector(lo, hi, 0,1,2,3,4,5,6,7,8,9,10,11,12,13,14,15);
}

static __device__ __forceinline__ v8f wmma_bf16(v16bf a, v16bf b, v8f c) {
  // 8 args: (neg_a, A, neg_b, B, c_mod, C, reuse_a, reuse_b)
  return __builtin_amdgcn_wmma_f32_16x16x32_bf16(false, a, false, b, (short)0, c,
                                                 false, false);
}

// ---------------------------------------------------------------- fp32 -> bf16
__global__ void __launch_bounds__(256) cvt_f32_bf16(const float* __restrict__ src,
                                                    __bf16* __restrict__ dst,
                                                    size_t n) {
  size_t i = (size_t)blockIdx.x * blockDim.x + threadIdx.x;
  size_t stride = (size_t)gridDim.x * blockDim.x;
  for (; i < n; i += stride) dst[i] = (__bf16)src[i];
}

// --------------------------------------------- C[M,N] f32 = A[M,K] * W[N,K]^T
// One wave computes a 64x64 output tile (4x4 WMMA fragments, 16 acc regs).
// Per K=32 step: 8 fragment loads feed 16 v_wmma -> 32 FLOP/byte intensity.
__global__ void __launch_bounds__(256) gemm_bf16_nt(const __bf16* __restrict__ A,
                                                    const __bf16* __restrict__ W,
                                                    float* __restrict__ C,
                                                    int M, int N, int K) {
  const int lane   = threadIdx.x & 31;
  const int tilesN = N >> 6;
  const int wid    = blockIdx.x * 8 + (threadIdx.x >> 5);
  const int tm     = wid / tilesN;
  const int tn     = wid - tm * tilesN;
  if (tm >= (M >> 6)) return;

  const int m0   = tm << 6;
  const int nn0  = tn << 6;
  const int rr   = lane & 15;             // A row / B row within 16-tile
  const int aoff = (lane < 16) ? 0 : 16;  // A K-offset for upper lane half
  const int boff = (lane < 16) ? 0 : 16;  // B K-offset for upper lane half

  v8f acc[4][4];
#pragma unroll
  for (int i = 0; i < 4; ++i)
#pragma unroll
    for (int j = 0; j < 4; ++j)
      acc[i][j] = (v8f){0.f,0.f,0.f,0.f,0.f,0.f,0.f,0.f};

  for (int k0 = 0; k0 < K; k0 += 32) {
    v16bf a[4], b[4];
#pragma unroll
    for (int i = 0; i < 4; ++i) {
      const __bf16* ap = A + (size_t)(m0 + 16 * i + rr) * K + k0;
      a[i] = cat8(*(const v8bf*)(ap + aoff), *(const v8bf*)(ap + 8 + aoff));
      __builtin_prefetch(ap + 128, 0, 3);   // next K-tiles -> global_prefetch_b8
    }
#pragma unroll
    for (int j = 0; j < 4; ++j) {
      const __bf16* wp = W + (size_t)(nn0 + 16 * j + rr) * K + k0 + boff;
      b[j] = *(const v16bf*)wp;
      __builtin_prefetch(wp + 128, 0, 3);
    }
#pragma unroll
    for (int i = 0; i < 4; ++i)
#pragma unroll
      for (int j = 0; j < 4; ++j)
        acc[i][j] = wmma_bf16(a[i], b[j], acc[i][j]);
  }

  const int mo = (lane < 16) ? 0 : 8;
#pragma unroll
  for (int i = 0; i < 4; ++i)
#pragma unroll
    for (int j = 0; j < 4; ++j)
#pragma unroll
      for (int r = 0; r < 8; ++r)
        C[(size_t)(m0 + 16 * i + r + mo) * N + nn0 + 16 * j + (lane & 15)] =
            acc[i][j][r];
}

// ----------------------- fused RMSNorm (full D) + RoPE + head-layout scatter
// q,k -> bf16 [B,H,L,hd] ; v -> bf16 transposed [B,H,hd,L]
__global__ void __launch_bounds__(256) normrope_kernel(
    const float* __restrict__ qkv, const float* __restrict__ freqs,
    const float* __restrict__ qw, const float* __restrict__ kw,
    __bf16* __restrict__ qh, __bf16* __restrict__ kh, __bf16* __restrict__ vt) {
  const int t   = threadIdx.x;
  const int gid = blockIdx.x;  // b*L + l
  const int b   = gid / L_;
  const int l   = gid - b * L_;
  const float* row = qkv + (size_t)gid * (3 * D_);
  const float* qr = row;
  const float* kr = row + D_;
  const float* vr = row + 2 * D_;

  float qe[4], qo[4], ke[4], ko[4];
  float qss = 0.f, kss = 0.f;
#pragma unroll
  for (int j = 0; j < 4; ++j) {
    int p = t + 256 * j;                  // pair index in [0, D/2)
    qe[j] = qr[2 * p]; qo[j] = qr[2 * p + 1];
    ke[j] = kr[2 * p]; ko[j] = kr[2 * p + 1];
    qss += qe[j] * qe[j] + qo[j] * qo[j];
    kss += ke[j] * ke[j] + ko[j] * ko[j];
  }
  __shared__ float sq[256], sk[256];
  sq[t] = qss; sk[t] = kss;
  __syncthreads();
  for (int s = 128; s > 0; s >>= 1) {
    if (t < s) { sq[t] += sq[t + s]; sk[t] += sk[t + s]; }
    __syncthreads();
  }
  const float qinv = rsqrtf(sq[0] * (1.0f / D_) + EPS_);
  const float kinv = rsqrtf(sk[0] * (1.0f / D_) + EPS_);

  const float* sinp = freqs + (size_t)l * (D_ / 2);
  const float* cosp = freqs + (size_t)L_ * (D_ / 2) + (size_t)l * (D_ / 2);

#pragma unroll
  for (int j = 0; j < 4; ++j) {
    int p = t + 256 * j;
    float s = sinp[p], c = cosp[p];
    int h = (2 * p) / HD_, d = (2 * p) % HD_;
    size_t base = ((size_t)(b * H_ + h) * L_ + l) * HD_ + d;

    float xe = qe[j] * qinv * qw[2 * p];
    float xo = qo[j] * qinv * qw[2 * p + 1];
    qh[base]     = (__bf16)(xe * c - xo * s);
    qh[base + 1] = (__bf16)(xo * c + xe * s);

    xe = ke[j] * kinv * kw[2 * p];
    xo = ko[j] * kinv * kw[2 * p + 1];
    kh[base]     = (__bf16)(xe * c - xo * s);
    kh[base + 1] = (__bf16)(xo * c + xe * s);
  }
#pragma unroll
  for (int j = 0; j < 8; ++j) {
    int i = t + 256 * j;
    vt[((size_t)(b * H_ + i / HD_) * HD_ + (i % HD_)) * L_ + l] = (__bf16)vr[i];
  }
}

// ------------------------------------------------ flash attention, bf16 WMMA
// One wave per 16 q-rows; block = 8 waves = 128 q-rows per (b,h) slice.
__global__ void __launch_bounds__(256) attn_kernel(const __bf16* __restrict__ qh,
                                                   const __bf16* __restrict__ kh,
                                                   const __bf16* __restrict__ vt,
                                                   __bf16* __restrict__ ob) {
  __shared__ __bf16 Pld[8][16 * 32];  // per-wave P staging (C-layout -> A-layout)

  const int wave  = threadIdx.x >> 5;
  const int lane  = threadIdx.x & 31;
  const int qtb   = blockIdx.x % (L_ / 128);
  const int bh    = blockIdx.x / (L_ / 128);
  const int b     = bh / H_;
  const int h     = bh - b * H_;
  const int qBase = qtb * 128 + wave * 16;

  const __bf16* qp = qh + (size_t)bh * L_ * HD_;
  const __bf16* kp = kh + (size_t)bh * L_ * HD_;
  const __bf16* vp = vt + (size_t)bh * HD_ * L_;  // [hd][L]

  const int n0   = lane & 15;
  const int mo   = (lane < 16) ? 0 : 8;
  const int aoff = (lane < 16) ? 0 : 16;
  const int boff = (lane < 16) ? 0 : 16;

  // Resident Q fragments: 16 x 128 over four K=32 steps
  v16bf qa[4];
#pragma unroll
  for (int ks = 0; ks < 4; ++ks) {
    const __bf16* ap = qp + (size_t)(qBase + n0) * HD_ + ks * 32;
    qa[ks] = cat8(*(const v8bf*)(ap + aoff), *(const v8bf*)(ap + 8 + aoff));
  }

  v8f o[8];
#pragma unroll
  for (int t = 0; t < 8; ++t) o[t] = (v8f){0.f,0.f,0.f,0.f,0.f,0.f,0.f,0.f};
  float mrow[8], lrow[8];
#pragma unroll
  for (int r = 0; r < 8; ++r) { mrow[r] = -1e30f; lrow[r] = 0.f; }

  const float scale = 0.08838834764831845f;  // 1/sqrt(128)

  for (int kt = 0; kt < L_; kt += 32) {
    // S (16x32) = Q @ K^T for this 32-key chunk
    v8f s0 = (v8f){0.f,0.f,0.f,0.f,0.f,0.f,0.f,0.f};
    v8f s1 = (v8f){0.f,0.f,0.f,0.f,0.f,0.f,0.f,0.f};
#pragma unroll
    for (int ks = 0; ks < 4; ++ks) {
      v16bf kb0 = *(const v16bf*)(kp + (size_t)(kt + n0) * HD_ + ks * 32 + boff);
      v16bf kb1 = *(const v16bf*)(kp + (size_t)(kt + 16 + n0) * HD_ + ks * 32 + boff);
      s0 = wmma_bf16(qa[ks], kb0, s0);
      s1 = wmma_bf16(qa[ks], kb1, s1);
    }
    // Online softmax in C-fragment layout: row m = r + mo lives on 16 lanes.
#pragma unroll
    for (int r = 0; r < 8; ++r) {
      float a = s0[r] * scale, c = s1[r] * scale;
      float rm = fmaxf(a, c);
      rm = fmaxf(rm, __shfl_xor(rm, 1, 16));
      rm = fmaxf(rm, __shfl_xor(rm, 2, 16));
      rm = fmaxf(rm, __shfl_xor(rm, 4, 16));
      rm = fmaxf(rm, __shfl_xor(rm, 8, 16));
      float mnew = fmaxf(mrow[r], rm);
      float corr = __expf(mrow[r] - mnew);
      float p0 = __expf(a - mnew);
      float p1 = __expf(c - mnew);
      float rs = p0 + p1;
      rs += __shfl_xor(rs, 1, 16);
      rs += __shfl_xor(rs, 2, 16);
      rs += __shfl_xor(rs, 4, 16);
      rs += __shfl_xor(rs, 8, 16);
      lrow[r] = lrow[r] * corr + rs;
      mrow[r] = mnew;
#pragma unroll
      for (int t = 0; t < 8; ++t) o[t][r] *= corr;
      Pld[wave][(r + mo) * 32 + n0]      = (__bf16)p0;
      Pld[wave][(r + mo) * 32 + 16 + n0] = (__bf16)p1;
    }
    asm volatile("s_wait_dscnt 0x0" ::: "memory");  // RAW on per-wave LDS tile
    // Re-read P as an A fragment (16x32)
    v16bf pa = cat8(*(const v8bf*)&Pld[wave][n0 * 32 + aoff],
                    *(const v8bf*)&Pld[wave][n0 * 32 + 8 + aoff]);
    // O += P @ V  (8 N-tiles of 16 over head dim)
#pragma unroll
    for (int t = 0; t < 8; ++t) {
      v16bf vb = *(const v16bf*)(vp + (size_t)(t * 16 + n0) * L_ + kt + boff);
      o[t] = wmma_bf16(pa, vb, o[t]);
    }
  }

  // Normalize and store to [B*L, D] bf16 for the output projection
#pragma unroll
  for (int r = 0; r < 8; ++r) {
    float inv = 1.0f / lrow[r];
    int row = qBase + r + mo;
#pragma unroll
    for (int t = 0; t < 8; ++t)
      ob[(size_t)(b * L_ + row) * D_ + h * HD_ + t * 16 + n0] =
          (__bf16)(o[t][r] * inv);
  }
}

// ----------------------------------------------------------------- launcher
extern "C" void kernel_launch(void* const* d_in, const int* in_sizes, int n_in,
                              void* d_out, int out_size, void* d_ws, size_t ws_size,
                              hipStream_t stream) {
  const float* x     = (const float*)d_in[0];
  const float* freqs = (const float*)d_in[1];
  const float* wqkv  = (const float*)d_in[2];
  const float* wo    = (const float*)d_in[3];
  const float* qw    = (const float*)d_in[4];
  const float* kw    = (const float*)d_in[5];
  float* out = (float*)d_out;

  char* ws = (char*)d_ws;
  size_t off = 0;
  auto take = [&](size_t bytes) -> char* {
    char* p = ws + off;
    off += (bytes + 255) & ~(size_t)255;
    return p;
  };

  const size_t BL = (size_t)B_ * L_;
  __bf16* xb    = (__bf16*)take(BL * D_ * 2);
  __bf16* wqkvb = (__bf16*)take((size_t)3 * D_ * D_ * 2);
  __bf16* wob   = (__bf16*)take((size_t)D_ * D_ * 2);
  float*  qkv   = (float*) take(BL * 3 * D_ * 4);
  __bf16* qhB   = (__bf16*)take(BL * D_ * 2);
  __bf16* khB   = (__bf16*)take(BL * D_ * 2);
  __bf16* vtB   = (__bf16*)take(BL * D_ * 2);
  __bf16* obB   = (__bf16*)take(BL * D_ * 2);

  cvt_f32_bf16<<<4096, 256, 0, stream>>>(x, xb, BL * D_);
  cvt_f32_bf16<<<4096, 256, 0, stream>>>(wqkv, wqkvb, (size_t)3 * D_ * D_);
  cvt_f32_bf16<<<2048, 256, 0, stream>>>(wo, wob, (size_t)D_ * D_);

  // qkv[8192, 6144] = x[8192, 2048] @ wqkv^T   (64x64 tiles per wave)
  gemm_bf16_nt<<<(8192 / 64) * (6144 / 64) / 8, 256, 0, stream>>>(
      xb, wqkvb, qkv, 8192, 6144, 2048);

  normrope_kernel<<<(int)BL, 256, 0, stream>>>(qkv, freqs, qw, kw, qhB, khB, vtB);

  attn_kernel<<<B_ * H_ * (L_ / 128), 256, 0, stream>>>(qhB, khB, vtB, obB);

  // out[8192, 2048] = o[8192, 2048] @ wo^T
  gemm_bf16_nt<<<(8192 / 64) * (2048 / 64) / 8, 256, 0, stream>>>(
      obB, wob, out, 8192, 2048, 2048);
}
